// BitLinear_14937896255490
// MI455X (gfx1250) — 8-bit matrix operand order repaired and checked by reading, compiled, not run on hardware
//
#include <hip/hip_runtime.h>
#include <hip/hip_bf16.h>
#include <stdint.h>

typedef __attribute__((ext_vector_type(8))) int v8i;

#define M_DIM 32768
#define K_DIM 512
#define N_DIM 2048
#define W_ELEMS (N_DIM * K_DIM)
#define EPSQ 1e-6f

#define QX_OFF   0u
#define QW_OFF   16777216u
#define SX_OFF   17825792u
#define PART_OFF 17956864u
#define SW_OFF   17957888u

__global__ __launch_bounds__(256) void wabs_partial(const float* __restrict__ w,
                                                    float* __restrict__ part) {
    __shared__ float red[256];
    const int CHUNK = W_ELEMS / 256;
    long base = (long)blockIdx.x * CHUNK;
    float s = 0.f;
    for (int i = threadIdx.x; i < CHUNK; i += 256)
        s += fabsf(w[base + i]);
    red[threadIdx.x] = s;
    __syncthreads();
    for (int off = 128; off; off >>= 1) {
        if (threadIdx.x < off) red[threadIdx.x] += red[threadIdx.x + off];
        __syncthreads();
    }
    if (threadIdx.x == 0) part[blockIdx.x] = red[0];
}

__global__ __launch_bounds__(256) void wabs_final(const float* __restrict__ part,
                                                  float* __restrict__ sw_out) {
    __shared__ float red[256];
    red[threadIdx.x] = part[threadIdx.x];
    __syncthreads();
    for (int off = 128; off; off >>= 1) {
        if (threadIdx.x < off) red[threadIdx.x] += red[threadIdx.x + off];
        __syncthreads();
    }
    if (threadIdx.x == 0) {
        float mean = red[0] / (float)W_ELEMS;
        sw_out[0] = fmaxf(mean, EPSQ);
    }
}

__global__ __launch_bounds__(256) void quant_w(const float* __restrict__ w,
                                               const float* __restrict__ sw_p,
                                               int8_t* __restrict__ qw) {
    float scale = 1.0f / sw_p[0];
    long i = ((long)blockIdx.x * 256 + threadIdx.x) * 4;
    float4 v = *reinterpret_cast<const float4*>(w + i);
    int q0 = max(-1, min(1, (int)rintf(v.x * scale)));
    int q1 = max(-1, min(1, (int)rintf(v.y * scale)));
    int q2 = max(-1, min(1, (int)rintf(v.z * scale)));
    int q3 = max(-1, min(1, (int)rintf(v.w * scale)));
    int packed = (q0 & 0xFF) | ((q1 & 0xFF) << 8) | ((q2 & 0xFF) << 16) | ((q3 & 0xFF) << 24);
    *reinterpret_cast<int*>(qw + i) = packed;
}

__global__ __launch_bounds__(256) void quant_act(const float* __restrict__ x,
                                                 int8_t* __restrict__ qx,
                                                 float* __restrict__ inv_sx) {
    int wave = threadIdx.x >> 5;
    int lane = threadIdx.x & 31;
    long row = (long)blockIdx.x * 8 + wave;
    const float* xr = x + row * K_DIM;

    float4 v[4];
    float amax = 0.f;
#pragma unroll
    for (int i = 0; i < 4; ++i) {
        v[i] = reinterpret_cast<const float4*>(xr)[lane * 4 + i];
        amax = fmaxf(amax, fmaxf(fmaxf(fabsf(v[i].x), fabsf(v[i].y)),
                                 fmaxf(fabsf(v[i].z), fabsf(v[i].w))));
    }
#pragma unroll
    for (int off = 16; off; off >>= 1)
        amax = fmaxf(amax, __shfl_xor(amax, off, 32));

    float m = fmaxf(amax, EPSQ);
    float scale = 127.0f / m;
    if (lane == 0) inv_sx[row] = m / 127.0f;

    int* qrow = reinterpret_cast<int*>(qx + row * K_DIM);
#pragma unroll
    for (int i = 0; i < 4; ++i) {
        int q0 = max(-128, min(127, (int)rintf(v[i].x * scale)));
        int q1 = max(-128, min(127, (int)rintf(v[i].y * scale)));
        int q2 = max(-128, min(127, (int)rintf(v[i].z * scale)));
        int q3 = max(-128, min(127, (int)rintf(v[i].w * scale)));
        qrow[lane * 4 + i] = (q0 & 0xFF) | ((q1 & 0xFF) << 8) |
                             ((q2 & 0xFF) << 16) | ((q3 & 0xFF) << 24);
    }
}

__global__ __launch_bounds__(256) void bitlinear_gemm_iu8(
        const int8_t* __restrict__ qx,
        const int8_t* __restrict__ qw,
        const float* __restrict__ inv_sx,
        const float* __restrict__ sw_p,
        const float* __restrict__ bias,
        float* __restrict__ out) {
    const int tilesN = N_DIM / 32;
    int wave = threadIdx.x >> 5;
    int lane = threadIdx.x & 31;
    int tile = blockIdx.x * 8 + wave;
    int tm = (tile / tilesN) * 32;
    int tn = (tile % tilesN) * 32;
    int h   = lane >> 4;
    int l15 = lane & 15;

    const int8_t* a0p = qx + (long)(tm + l15) * K_DIM;
    const int8_t* a1p = a0p + 16 * K_DIM;
    const int8_t* b0p = qw + (long)(tn + l15) * K_DIM;
    const int8_t* b1p = b0p + 16 * K_DIM;

    v8i acc00 = {}, acc01 = {}, acc10 = {}, acc11 = {};
    for (int k0 = 0; k0 < K_DIM; k0 += 64) {
        v8i a0, a1, b0, b1;
#pragma unroll
        for (int v = 0; v < 8; ++v) {
            int ka = k0 + ((v >> 1) << 4) + ((v & 1) << 2) + (h << 3);
            a0[v] = *reinterpret_cast<const int*>(a0p + ka);
            a1[v] = *reinterpret_cast<const int*>(a1p + ka);
            int kb = k0 + ((v >> 1) << 4) + ((v & 1) << 2) + (h << 3);
            b0[v] = *reinterpret_cast<const int*>(b0p + kb);
            b1[v] = *reinterpret_cast<const int*>(b1p + kb);
        }
        acc00 = __builtin_amdgcn_wmma_i32_16x16x64_iu8(true, a0, true, b0, acc00, false, false);
        acc01 = __builtin_amdgcn_wmma_i32_16x16x64_iu8(true, a0, true, b1, acc01, false, false);
        acc10 = __builtin_amdgcn_wmma_i32_16x16x64_iu8(true, a1, true, b0, acc10, false, false);
        acc11 = __builtin_amdgcn_wmma_i32_16x16x64_iu8(true, a1, true, b1, acc11, false, false);
    }

    float sw  = sw_p[0];
    float fb0 = bias[tn + l15];
    float fb1 = bias[tn + 16 + l15];
#pragma unroll
    for (int r = 0; r < 8; ++r) {
        int m0 = tm + r + (h << 3);
        int m1 = m0 + 16;
        float s0 = inv_sx[m0] * sw;
        float s1 = inv_sx[m1] * sw;
        long o0 = (long)m0 * N_DIM + tn + l15;
        long o1 = (long)m1 * N_DIM + tn + l15;
        out[o0]      = (float)acc00[r] * s0 + fb0;
        out[o0 + 16] = (float)acc01[r] * s0 + fb1;
        out[o1]      = (float)acc10[r] * s1 + fb0;
        out[o1 + 16] = (float)acc11[r] * s1 + fb1;
    }
}

extern "C" void kernel_launch(void* const* d_in, const int* in_sizes, int n_in,
                              void* d_out, int out_size, void* d_ws, size_t ws_size,
                              hipStream_t stream) {
    const float* x    = (const float*)d_in[0];
    const float* w    = (const float*)d_in[1];
    const float* bias = (const float*)d_in[2];
    float* out = (float*)d_out;

    uint8_t* ws = (uint8_t*)d_ws;
    int8_t* qx    = (int8_t*)(ws + QX_OFF);
    int8_t* qw    = (int8_t*)(ws + QW_OFF);
    float*  sx    = (float*)(ws + SX_OFF);
    float*  part  = (float*)(ws + PART_OFF);
    float*  sw    = (float*)(ws + SW_OFF);

    wabs_partial<<<256, 256, 0, stream>>>(w, part);
    wabs_final<<<1, 256, 0, stream>>>(part, sw);

    quant_w<<<1024, 256, 0, stream>>>(w, sw, qw);

    quant_act<<<4096, 256, 0, stream>>>(x, qx, sx);

    bitlinear_gemm_iu8<<<8192, 256, 0, stream>>>(qx, qw, sx, sw, bias, out);
}
